// Attention_48876727828718
// MI455X (gfx1250) — matrix operand order repaired and checked by reading, not compiled, not run on hardware
//
#include <hip/hip_runtime.h>
#include <hip/hip_bf16.h>

#define BB 2
#define SS 2048
#define EE 64
#define HH 8

typedef __bf16 bf16;
typedef __attribute__((ext_vector_type(8)))  bf16  v8bf;
typedef __attribute__((ext_vector_type(16))) bf16  v16bf;
typedef __attribute__((ext_vector_type(8)))  float v8f;
typedef __attribute__((ext_vector_type(4)))  float v4f;

static __device__ __forceinline__ v8f wmma_bf16(v16bf a, v16bf b, v8f c) {
  return __builtin_amdgcn_wmma_f32_16x16x32_bf16(false, a, false, b, (short)0, c, false, false);
}

// A fragment: row-major [16 x K] bf16, element (m,k) at base[m*lda + k].
// lanes 0-15: K kbase+0..15 ; lanes 16-31: K kbase+16..31
static __device__ __forceinline__ v16bf load_A(const bf16* base, int lda, int kbase, int lane) {
  const bf16* p = base + (lane & 15) * lda + kbase + ((lane >> 4) << 4);
  v8bf lo = *(const v8bf*)(p);
  v8bf hi = *(const v8bf*)(p + 8);
  v16bf r;
#pragma unroll
  for (int i = 0; i < 8; ++i) { r[i] = lo[i]; r[8 + i] = hi[i]; }
  return r;
}

// B fragment: element (k,n) at base[n*colStride + k], K=0..31, N=0..15.
static __device__ __forceinline__ v16bf load_B(const bf16* base, int colStride, int lane) {
  const bf16* p = base + (lane & 15) * colStride + ((lane >> 4) << 4);
  v8bf lo = *(const v8bf*)(p);
  v8bf hi = *(const v8bf*)(p + 8);
  v16bf r;
#pragma unroll
  for (int i = 0; i < 8; ++i) { r[i] = lo[i]; r[8 + i] = hi[i]; }
  return r;
}

// Pre-packed B fragment: frag stored as 32 lanes x 16 bf16, contiguous per lane.
static __device__ __forceinline__ v16bf load_pack(const bf16* pk, int frag, int lane) {
  const bf16* p = pk + frag * 512 + lane * 16;
  v8bf lo = *(const v8bf*)(p);
  v8bf hi = *(const v8bf*)(p + 8);
  v16bf r;
#pragma unroll
  for (int i = 0; i < 8; ++i) { r[i] = lo[i]; r[8 + i] = hi[i]; }
  return r;
}

// ---------------- Kernel 0: pack fp32 weights into WMMA B-fragment bf16 layout ----------------
// Blocks 0..191  : projection fragments. f = m*64 + h*8 + ft*2 + c
//                  (m: 0=Wq,1=Wk,2=Wv ; ft: N-tile ; c: K-chunk of 32)
// Blocks 192..255: W_o fragments. g = t*16 + kc  (t: N-tile ; kc: K-chunk of 32)
__global__ void pack_kernel(const float* __restrict__ Wq, const float* __restrict__ Wk,
                            const float* __restrict__ Wv, const float* __restrict__ Wo,
                            bf16* __restrict__ wpack, bf16* __restrict__ wopack) {
  const int lane = threadIdx.x;
  const int f = blockIdx.x;
  const int n = lane & 15;
  if (f < 192) {
    const int m = f / 64;
    const int rem = f % 64;
    const int h = rem / 8, ft = (rem % 8) / 2, c = rem % 2;
    const float* W = (m == 0 ? Wq : (m == 1 ? Wk : Wv)) + h * EE * EE;
    const int k0 = c * 32 + ((lane >> 4) << 4);
    bf16* dst = wpack + f * 512 + lane * 16;
#pragma unroll
    for (int i = 0; i < 16; ++i) dst[i] = (bf16)W[(k0 + i) * EE + ft * 16 + n];
  } else {
    const int g = f - 192;
    const int t = g / 16, kc = g % 16;
    const int k0 = kc * 32 + ((lane >> 4) << 4);
    bf16* dst = wopack + g * 512 + lane * 16;
#pragma unroll
    for (int i = 0; i < 16; ++i) dst[i] = (bf16)Wo[(t * 16 + n) * (HH * EE) + k0 + i];
  }
}

// ---------------- Kernel 1: LayerNorm -> bf16 xn ----------------
__global__ void ln_kernel(const float* __restrict__ x, const float* __restrict__ w,
                          bf16* __restrict__ xn) {
  const int row = blockIdx.x, lane = threadIdx.x;
  const float* xr = x + row * EE;
  float a = xr[lane], c = xr[lane + 32];
  float s = a + c;
#pragma unroll
  for (int off = 16; off >= 1; off >>= 1) s += __shfl_xor(s, off, 32);
  const float mu = s * (1.0f / EE);
  const float da = a - mu, dc = c - mu;
  float v = da * da + dc * dc;
#pragma unroll
  for (int off = 16; off >= 1; off >>= 1) v += __shfl_xor(v, off, 32);
  const float rstd = rsqrtf(v * (1.0f / EE) + 1e-5f);
  xn[row * EE + lane]      = (bf16)(da * rstd * w[lane]);
  xn[row * EE + lane + 32] = (bf16)(dc * rstd * w[lane + 32]);
}

// ---------------- Kernel 2: per-head Q/K/V projections + row norms ----------------
// Each wave covers 4 row tiles (64 rows). Three sequential passes (Q, K, V), weight
// fragments loaded from the packed buffer (2 contiguous 16B loads each, no cvt).
#define PT 4
__global__ void __launch_bounds__(32, 1)
proj_kernel(const bf16* __restrict__ xn, const bf16* __restrict__ wpack,
            bf16* __restrict__ Qb, bf16* __restrict__ Kb, bf16* __restrict__ Vt,
            float* __restrict__ q2, float* __restrict__ k2) {
  const int lane = threadIdx.x;
  const int nseg = SS / (16 * PT);             // 32 segments per (b,h)
  const int seg = blockIdx.x % nseg;
  const int bh  = blockIdx.x / nseg;
  const int b   = bh / HH, h = bh % HH;
  const int row0 = seg * 16 * PT;

  const int n  = lane & 15;
  const int mb = (lane >> 4) << 3;

#pragma unroll
  for (int m = 0; m < 3; ++m) {   // 0=Q, 1=K, 2=V
    v16bf wb[4][2];
#pragma unroll
    for (int ft = 0; ft < 4; ++ft) {
      wb[ft][0] = load_pack(wpack, m * 64 + h * 8 + ft * 2 + 0, lane);
      wb[ft][1] = load_pack(wpack, m * 64 + h * 8 + ft * 2 + 1, lane);
    }
    for (int t = 0; t < PT; ++t) {
      const int r0 = row0 + t * 16;
      const bf16* xp = xn + (b * SS + r0) * EE;
      const v16bf a0 = load_A(xp, EE, 0, lane);
      const v16bf a1 = load_A(xp, EE, 32, lane);
      float sq[8];
#pragma unroll
      for (int r = 0; r < 8; ++r) sq[r] = 0.0f;
#pragma unroll
      for (int ft = 0; ft < 4; ++ft) {
        v8f c = {};
        c = wmma_bf16(a0, wb[ft][0], c);
        c = wmma_bf16(a1, wb[ft][1], c);
        if (m == 2) {  // V: transposed [e][s] store (contiguous per lane -> b128)
#pragma unroll
          for (int r = 0; r < 8; ++r)
            Vt[(bh * EE + ft * 16 + n) * SS + r0 + mb + r] = (bf16)c[r];
        } else {
          bf16* dst = (m == 0 ? Qb : Kb);
#pragma unroll
          for (int r = 0; r < 8; ++r) {
            dst[(bh * SS + r0 + mb + r) * EE + ft * 16 + n] = (bf16)c[r];
            sq[r] += c[r] * c[r];
          }
        }
      }
      if (m != 2) {
#pragma unroll
        for (int r = 0; r < 8; ++r) {
#pragma unroll
          for (int off = 8; off >= 1; off >>= 1) sq[r] += __shfl_xor(sq[r], off, 32);
        }
        if (n == 0) {
          float* nd = (m == 0 ? q2 : k2);
#pragma unroll
          for (int r = 0; r < 8; ++r) nd[bh * SS + r0 + mb + r] = sq[r];
        }
      }
    }
  }
}

// ---------------- Kernel 3: flash-style causal RBF attention ----------------
// Each wave handles 32 query rows (2 tiles); K/V fragments are loaded once per
// 32-key block and reused by both query tiles.
__global__ void __launch_bounds__(32, 1)
attn_kernel(const bf16* __restrict__ Qb, const bf16* __restrict__ Kb,
            const bf16* __restrict__ Vt,
            const float* __restrict__ q2, const float* __restrict__ k2,
            const float* __restrict__ gamma, bf16* __restrict__ cc) {
  __shared__ float sc[2 * 16 * 34];
  const int lane = threadIdx.x;
  const int ntile = SS / 32;
  const int qt = blockIdx.x % ntile;
  const int bh = blockIdx.x / ntile;
  const int b  = bh / HH, h = bh % HH;
  const int q0 = qt * 32;

  const bf16* Qp = Qb + bh * SS * EE;
  const bf16* Kp = Kb + bh * SS * EE;
  const bf16* Vp = Vt + bh * EE * SS;
  const float* q2p = q2 + bh * SS;
  const float* k2p = k2 + bh * SS;
  const float gsc = gamma[h] * 0.125f;  // gamma / sqrt(E)

  const int n  = lane & 15;
  const int mb = (lane >> 4) << 3;

  v16bf aq[2][2];
  float q2v[2][8];
#pragma unroll
  for (int u = 0; u < 2; ++u) {
    aq[u][0] = load_A(Qp + (q0 + u * 16) * EE, EE, 0, lane);
    aq[u][1] = load_A(Qp + (q0 + u * 16) * EE, EE, 32, lane);
    const v4f lo = *(const v4f*)(q2p + q0 + u * 16 + mb);
    const v4f hi = *(const v4f*)(q2p + q0 + u * 16 + mb + 4);
#pragma unroll
    for (int i = 0; i < 4; ++i) { q2v[u][i] = lo[i]; q2v[u][4 + i] = hi[i]; }
  }

  v8f acc[2][4];
#pragma unroll
  for (int u = 0; u < 2; ++u)
#pragma unroll
    for (int t = 0; t < 4; ++t) acc[u][t] = (v8f){};

  const int jend = q0 + 32;
  for (int j0 = 0; j0 < jend; j0 += 32) {
    __builtin_prefetch(Kp + (j0 + 32) * EE, 0, 1);
    // K^T B fragments for keys j0..j0+31 (two 16-key tiles x two 32-e chunks)
    const v16bf kb00 = load_B(Kp + j0 * EE,             EE, lane);
    const v16bf kb01 = load_B(Kp + j0 * EE + 32,        EE, lane);
    const v16bf kb10 = load_B(Kp + (j0 + 16) * EE,      EE, lane);
    const v16bf kb11 = load_B(Kp + (j0 + 16) * EE + 32, EE, lane);
    const float k2a = k2p[j0 + n];
    const float k2b = k2p[j0 + 16 + n];

#pragma unroll
    for (int u = 0; u < 2; ++u) {
      v8f s0 = {}, s1 = {};
      s0 = wmma_bf16(aq[u][0], kb00, s0);
      s0 = wmma_bf16(aq[u][1], kb01, s0);
      s1 = wmma_bf16(aq[u][0], kb10, s1);
      s1 = wmma_bf16(aq[u][1], kb11, s1);
      const int base = q0 + u * 16;
      float* scr = sc + u * (16 * 34);
#pragma unroll
      for (int r = 0; r < 8; ++r) {
        const int m = mb + r;
        const float d0 = q2v[u][r] + k2a - 2.0f * s0[r];
        scr[m * 34 + n]      = (j0 + n      <= base + m) ? __expf(-gsc * d0) : 0.0f;
        const float d1 = q2v[u][r] + k2b - 2.0f * s1[r];
        scr[m * 34 + 16 + n] = (j0 + 16 + n <= base + m) ? __expf(-gsc * d1) : 0.0f;
      }
    }
    __syncthreads();
    // Re-layout both 16x32 fp32 score tiles (C-layout) into bf16 A fragments
    v16bf as[2];
    {
      const int mm = lane & 15;
      const int kh = (lane >> 4) << 4;
#pragma unroll
      for (int u = 0; u < 2; ++u) {
        const float* scr = sc + u * (16 * 34);
#pragma unroll
        for (int i = 0; i < 8; ++i) {
          as[u][i]     = (bf16)scr[mm * 34 + kh + i];
          as[u][8 + i] = (bf16)scr[mm * 34 + kh + 8 + i];
        }
      }
    }
    __syncthreads();
    // O += scores(16x32) x V(32x64); V fragments shared by both query tiles
#pragma unroll
    for (int t = 0; t < 4; ++t) {
      const v16bf vb = load_B(Vp + t * 16 * SS + j0, SS, lane);
      acc[0][t] = wmma_bf16(as[0], vb, acc[0][t]);
      acc[1][t] = wmma_bf16(as[1], vb, acc[1][t]);
    }
  }

  // Store into concat buffer [B, S, H*E] as bf16 (feeds the W_o matmul)
#pragma unroll
  for (int u = 0; u < 2; ++u) {
    bf16* cp = cc + (b * SS + q0 + u * 16) * (HH * EE) + h * EE;
#pragma unroll
    for (int r = 0; r < 8; ++r) {
      bf16* row = cp + (mb + r) * (HH * EE);
      row[0 * 16 + n] = (bf16)acc[u][0][r];
      row[1 * 16 + n] = (bf16)acc[u][1][r];
      row[2 * 16 + n] = (bf16)acc[u][2][r];
      row[3 * 16 + n] = (bf16)acc[u][3][r];
    }
  }
}

// ---------------- Kernel 4: out = concat @ W_o^T ----------------
__global__ void __launch_bounds__(32, 1)
out_kernel(const bf16* __restrict__ cc, const bf16* __restrict__ wopack,
           float* __restrict__ out) {
  const int lane = threadIdx.x;
  const int row0 = blockIdx.x * 16;
  const bf16* ap = cc + row0 * (HH * EE);
  const int n  = lane & 15;
  const int mb = (lane >> 4) << 3;

  v8f acc0 = {}, acc1 = {}, acc2 = {}, acc3 = {};
#pragma unroll
  for (int kc = 0; kc < 16; ++kc) {
    const v16bf a = load_A(ap, HH * EE, kc * 32, lane);
    acc0 = wmma_bf16(a, load_pack(wopack, 0 * 16 + kc, lane), acc0);
    acc1 = wmma_bf16(a, load_pack(wopack, 1 * 16 + kc, lane), acc1);
    acc2 = wmma_bf16(a, load_pack(wopack, 2 * 16 + kc, lane), acc2);
    acc3 = wmma_bf16(a, load_pack(wopack, 3 * 16 + kc, lane), acc3);
  }
#pragma unroll
  for (int r = 0; r < 8; ++r) {
    float* row = out + (row0 + mb + r) * EE;
    row[0 * 16 + n] = acc0[r];
    row[1 * 16 + n] = acc1[r];
    row[2 * 16 + n] = acc2[r];
    row[3 * 16 + n] = acc3[r];
  }
}

extern "C" void kernel_launch(void* const* d_in, const int* in_sizes, int n_in,
                              void* d_out, int out_size, void* d_ws, size_t ws_size,
                              hipStream_t stream) {
  const float* x     = (const float*)d_in[0];
  const float* ln_w  = (const float*)d_in[1];
  const float* Wq    = (const float*)d_in[2];
  const float* Wk    = (const float*)d_in[3];
  const float* Wv    = (const float*)d_in[4];
  const float* Wo    = (const float*)d_in[5];
  const float* gamma = (const float*)d_in[6];

  char* ws = (char*)d_ws;
  const size_t sz_xn  = (size_t)BB * SS * EE * sizeof(bf16);          // 512 KB
  const size_t sz_qkv = (size_t)BB * HH * SS * EE * sizeof(bf16);     // 4 MB each
  const size_t sz_n   = (size_t)BB * HH * SS * sizeof(float);         // 128 KB each
  const size_t sz_cc  = (size_t)BB * SS * HH * EE * sizeof(bf16);     // 4 MB
  const size_t sz_wp  = (size_t)192 * 512 * sizeof(bf16);             // 192 KB
  bf16* xn = (bf16*)(ws);
  bf16* Qb = (bf16*)(ws + sz_xn);
  bf16* Kb = (bf16*)(ws + sz_xn + sz_qkv);
  bf16* Vt = (bf16*)(ws + sz_xn + 2 * sz_qkv);
  float* q2 = (float*)(ws + sz_xn + 3 * sz_qkv);
  float* k2 = (float*)(ws + sz_xn + 3 * sz_qkv + sz_n);
  bf16* cc  = (bf16*)(ws + sz_xn + 3 * sz_qkv + 2 * sz_n);
  bf16* wpack  = (bf16*)(ws + sz_xn + 3 * sz_qkv + 2 * sz_n + sz_cc);
  bf16* wopack = (bf16*)(ws + sz_xn + 3 * sz_qkv + 2 * sz_n + sz_cc + sz_wp);

  pack_kernel<<<256, 32, 0, stream>>>(Wq, Wk, Wv, Wo, wpack, wopack);
  ln_kernel<<<BB * SS, 32, 0, stream>>>(x, ln_w, xn);
  proj_kernel<<<BB * HH * (SS / (16 * PT)), 32, 0, stream>>>(xn, wpack, Qb, Kb, Vt, q2, k2);
  attn_kernel<<<BB * HH * (SS / 32), 32, 0, stream>>>(Qb, Kb, Vt, q2, k2, gamma, cc);
  out_kernel<<<(BB * SS) / 16, 32, 0, stream>>>(cc, wopack, (float*)d_out);
}
